// PreMLP_70583492543028
// MI455X (gfx1250) — matrix operand order repaired and checked by reading, not compiled, not run on hardware
//
#include <hip/hip_runtime.h>
#include <hip/hip_bf16.h>

typedef __attribute__((ext_vector_type(16))) _Float16 v16h;
typedef __attribute__((ext_vector_type(8)))  _Float16 v8h;
typedef __attribute__((ext_vector_type(8)))  float    v8f;

// ---------------------------------------------------------------------------
// Farthest point sampling: one block per batch, 1024 threads (32 waves).
// ---------------------------------------------------------------------------
__global__ __launch_bounds__(1024)
void k_fps(const float* __restrict__ xyz, float* __restrict__ newxyz,
           float* __restrict__ dist, int N, int np) {
  const int b = blockIdx.x;
  const float* P = xyz + (size_t)b * N * 3;
  float* D = dist + (size_t)b * N;
  float* O = newxyz + (size_t)b * np * 3;
  const int t = threadIdx.x, nt = blockDim.x;

  __shared__ float s_val[1024];
  __shared__ int   s_idx[1024];
  __shared__ int   s_sel;

  for (int i = t; i < N; i += nt) D[i] = 1e10f;
  if (t == 0) s_sel = 0;
  __syncthreads();

  for (int it = 0; it < np; ++it) {
    const int sel = s_sel;
    const float cx = P[sel * 3 + 0], cy = P[sel * 3 + 1], cz = P[sel * 3 + 2];
    if (t == 0) { O[it * 3 + 0] = cx; O[it * 3 + 1] = cy; O[it * 3 + 2] = cz; }
    float best = -1.0f; int bi = 0;
    for (int i = t; i < N; i += nt) {
      const float dx = P[i * 3 + 0] - cx;
      const float dy = P[i * 3 + 1] - cy;
      const float dz = P[i * 3 + 2] - cz;
      const float d = dx * dx + dy * dy + dz * dz;
      float dd = D[i];
      if (d < dd) { dd = d; D[i] = d; }
      if (dd > best) { best = dd; bi = i; }
    }
    s_val[t] = best; s_idx[t] = bi;
    __syncthreads();
    for (int off = nt >> 1; off > 0; off >>= 1) {
      if (t < off) {
        const float ov = s_val[t + off]; const int oi = s_idx[t + off];
        if (ov > s_val[t] || (ov == s_val[t] && oi < s_idx[t])) { s_val[t] = ov; s_idx[t] = oi; }
      }
      __syncthreads();
    }
    if (t == 0) s_sel = s_idx[0];
    __syncthreads();
  }
}

// ---------------------------------------------------------------------------
// Ball query: first `ns` in-radius indices in ascending order (ref semantics).
// ---------------------------------------------------------------------------
__global__ __launch_bounds__(256)
void k_ballquery(const float* __restrict__ xyz, const float* __restrict__ nxyz,
                 int* __restrict__ idx, int B, int N, int np, int ns, float r2) {
  const int p = blockIdx.x * blockDim.x + threadIdx.x;
  if (p >= B * np) return;
  const int b = p / np;
  const float* P = xyz + (size_t)b * N * 3;
  const float qx = nxyz[p * 3 + 0], qy = nxyz[p * 3 + 1], qz = nxyz[p * 3 + 2];
  int* o = idx + (size_t)p * ns;
  int cnt = 0, first = -1;
  for (int j = 0; j < N && cnt < ns; ++j) {
    const float dx = P[j * 3 + 0] - qx;
    const float dy = P[j * 3 + 1] - qy;
    const float dz = P[j * 3 + 2] - qz;
    if (dx * dx + dy * dy + dz * dz <= r2) { if (first < 0) first = j; o[cnt++] = j; }
  }
  if (first < 0) first = 0;
  while (cnt < ns) o[cnt++] = first;
}

// ---------------------------------------------------------------------------
// Gather grouped inputs to f16: [ (xyz-center)/r | feats(f16) ], zero padded.
// ---------------------------------------------------------------------------
__global__ __launch_bounds__(256)
void k_gather(const float* __restrict__ xyz, const _Float16* __restrict__ feats,
              const float* __restrict__ nxyz, const int* __restrict__ idx,
              _Float16* __restrict__ X, int B, int N, int np, int ns, int C,
              int Kpad, float inv_r) {
  const int row = blockIdx.x * blockDim.x + threadIdx.x;
  const int total = B * np * ns;
  if (row >= total) return;
  const int p = row / ns;
  const int b = p / np;
  const int j = idx[row];
  const float* P = xyz + ((size_t)b * N + j) * 3;
  const float* Q = nxyz + (size_t)p * 3;
  _Float16* o = X + (size_t)row * Kpad;
  o[0] = (_Float16)((P[0] - Q[0]) * inv_r);
  o[1] = (_Float16)((P[1] - Q[1]) * inv_r);
  o[2] = (_Float16)((P[2] - Q[2]) * inv_r);
  int c = 3;
  if (feats) {
    const _Float16* F = feats + ((size_t)b * N + j) * C;
    for (int k = 0; k < C; ++k) o[3 + k] = F[k];
    c = 3 + C;
  }
  for (; c < Kpad; ++c) o[c] = (_Float16)0.0f;
}

// ---------------------------------------------------------------------------
// Weight conversion: f32 (N x K) -> f16 (Npad x Kpad), zero padded.
// ---------------------------------------------------------------------------
__global__ __launch_bounds__(256)
void k_cvt_w(const float* __restrict__ src, _Float16* __restrict__ dst,
             int N, int Npad, int K, int Kpad) {
  const int i = blockIdx.x * blockDim.x + threadIdx.x;
  if (i >= Npad * Kpad) return;
  const int n = i / Kpad, k = i % Kpad;
  dst[i] = (n < N && k < K) ? (_Float16)src[(size_t)n * K + k] : (_Float16)0.0f;
}

__global__ __launch_bounds__(256)
void k_pad_vec(const float* __restrict__ s, float* __restrict__ d, int n, int pad) {
  const int i = blockIdx.x * blockDim.x + threadIdx.x;
  if (i < pad) d[i] = (i < n) ? s[i] : 0.0f;
}

// ---------------------------------------------------------------------------
// WMMA GEMM: Y[M,N](f16) = relu( (X[M,K]f16 * W[N,K]^T f16) * s[n] + b[n] )
// Wave32 fragments per CDNA5 ISA 7.12.2. Tile shape is a compile-time
// template so the K-loop is straight-line: MSUB*NSUB WMMAs per K-step, no
// conditional accumulator updates (avoids phi/select register copies).
// Each B fragment is reused across MSUB M-tiles; each A fragment across
// NSUB N-tiles. M%16==0, K%32==0, N%(16*NSUB)==0 handled by launcher.
// ---------------------------------------------------------------------------
template <int MSUB, int NSUB>
__global__ __launch_bounds__(256)
void k_gemm_relu(const _Float16* __restrict__ X, const _Float16* __restrict__ W,
                 const float* __restrict__ S, const float* __restrict__ Bv,
                 _Float16* __restrict__ Y, int M, int N, int K) {
  const int wave = threadIdx.x >> 5;
  const int lane = threadIdx.x & 31;
  const int lm = lane & 15;       // M row within tile / N col within subtile
  const int lh = lane >> 4;       // lane half selects K sub-block
  const int tm0 = (blockIdx.y * 8 + wave) * MSUB;  // first M tile of this wave
  if (tm0 * 16 >= M) return;                       // wave-uniform
  const int nt0 = blockIdx.x * NSUB;               // first N subtile

  const _Float16* xr[MSUB];
#pragma unroll
  for (int mi = 0; mi < MSUB; ++mi)
    xr[mi] = X + (size_t)((tm0 + mi) * 16 + lm) * K;
  const _Float16* wr0 = W + (size_t)(nt0 * 16 + lm) * K;

  v8f acc[MSUB][NSUB];
#pragma unroll
  for (int mi = 0; mi < MSUB; ++mi)
#pragma unroll
    for (int s = 0; s < NSUB; ++s) {
      const v8f z = {0.f, 0.f, 0.f, 0.f, 0.f, 0.f, 0.f, 0.f};
      acc[mi][s] = z;
    }

  for (int kb = 0; kb < K; kb += 32) {
    v16h a[MSUB];
#pragma unroll
    for (int mi = 0; mi < MSUB; ++mi) {
      // A fragment (16x32 f16): halves 0..7 = K kb+8*lh.., halves 8..15 = +16.
      const v8h alo = *(const v8h*)(xr[mi] + kb + 8 * lh);
      const v8h ahi = *(const v8h*)(xr[mi] + kb + 16 + 8 * lh);
      a[mi] = __builtin_shufflevector(alo, ahi, 0, 1, 2, 3, 4, 5, 6, 7,
                                      8, 9, 10, 11, 12, 13, 14, 15);
      __builtin_prefetch(xr[mi] + kb + 32, 0, 1);  // global_prefetch_b8
    }
#pragma unroll
    for (int s = 0; s < NSUB; ++s) {
      // B fragment of W^T: lane lm = column lm of subtile (= row of W),
      // halves 0..7 = K rows kb+8*lh.., halves 8..15 = +16.
      const v8h blo = *(const v8h*)(wr0 + (size_t)s * 16 * K + kb + 8 * lh);
      const v8h bhi = *(const v8h*)(wr0 + (size_t)s * 16 * K + kb + 16 + 8 * lh);
      const v16h bfr = __builtin_shufflevector(blo, bhi, 0, 1, 2, 3, 4, 5, 6, 7,
                                               8, 9, 10, 11, 12, 13, 14, 15);
#pragma unroll
      for (int mi = 0; mi < MSUB; ++mi)
        acc[mi][s] = __builtin_amdgcn_wmma_f32_16x16x32_f16(
            false, a[mi], false, bfr, (short)0, acc[mi][s], false, false);
    }
  }

#pragma unroll
  for (int s = 0; s < NSUB; ++s) {
    const int n = (nt0 + s) * 16 + lm;
    const float sc = S  ? S[n]  : 1.0f;
    const float bb = Bv ? Bv[n] : 0.0f;
#pragma unroll
    for (int mi = 0; mi < MSUB; ++mi) {
#pragma unroll
      for (int r = 0; r < 8; ++r) {
        const int row = (tm0 + mi) * 16 + r + 8 * lh;  // D: M = r + 8*(lane/16)
        float v = acc[mi][s][r] * sc + bb;
        v = v > 0.0f ? v : 0.0f;
        Y[(size_t)row * N + n] = (_Float16)v;
      }
    }
  }
}

// ---------------------------------------------------------------------------
// Max pool over the nsample axis: (B*np, ns, C) -> (B*np, C), f16.
// ---------------------------------------------------------------------------
__global__ __launch_bounds__(256)
void k_maxpool(const _Float16* __restrict__ Y, _Float16* __restrict__ out,
               int total, int ns, int C) {
  const int o = blockIdx.x * blockDim.x + threadIdx.x;
  if (o >= total) return;
  const int c = o % C;
  const size_t p = (size_t)(o / C);
  const _Float16* y = Y + p * ns * (size_t)C + c;
  float m = (float)y[0];
  for (int k = 1; k < ns; ++k) { const float v = (float)y[(size_t)k * C]; m = v > m ? v : m; }
  out[o] = (_Float16)m;
}

// ---------------------------------------------------------------------------
// FP front-end: 3-NN inverse-distance interpolation + concat (f16 in/out).
// ---------------------------------------------------------------------------
__global__ __launch_bounds__(256)
void k_fp_interp(const float* __restrict__ xyz1, const float* __restrict__ xyz2,
                 const _Float16* __restrict__ f1, const _Float16* __restrict__ f2,
                 _Float16* __restrict__ X, int B, int n1, int n2, int C) {
  const int p = blockIdx.x * blockDim.x + threadIdx.x;
  if (p >= B * n1) return;
  const int b = p / n1;
  const float qx = xyz1[p * 3 + 0], qy = xyz1[p * 3 + 1], qz = xyz1[p * 3 + 2];
  const float* P2 = xyz2 + (size_t)b * n2 * 3;
  float d0 = 1e30f, d1 = 1e30f, d2 = 1e30f;
  int i0 = 0, i1 = 0, i2 = 0;
  for (int j = 0; j < n2; ++j) {
    const float dx = P2[j * 3 + 0] - qx;
    const float dy = P2[j * 3 + 1] - qy;
    const float dz = P2[j * 3 + 2] - qz;
    const float d = dx * dx + dy * dy + dz * dz;
    if (d < d0)      { d2 = d1; i2 = i1; d1 = d0; i1 = i0; d0 = d; i0 = j; }
    else if (d < d1) { d2 = d1; i2 = i1; d1 = d;  i1 = j; }
    else if (d < d2) { d2 = d;  i2 = j; }
  }
  float w0 = 1.0f / (d0 + 1e-8f), w1 = 1.0f / (d1 + 1e-8f), w2 = 1.0f / (d2 + 1e-8f);
  const float ws = w0 + w1 + w2;
  w0 /= ws; w1 /= ws; w2 /= ws;
  const _Float16* F2 = f2 + (size_t)b * n2 * C;
  const _Float16* F1 = f1 + (size_t)p * C;
  _Float16* o = X + (size_t)p * (2 * C);
  for (int c = 0; c < C; ++c) {
    const float v = w0 * (float)F2[(size_t)i0 * C + c] +
                    w1 * (float)F2[(size_t)i1 * C + c] +
                    w2 * (float)F2[(size_t)i2 * C + c];
    o[c] = (_Float16)v;
    o[C + c] = F1[c];
  }
}

// Head: pack column 0 of Y1 (B*1024 x 16, f16) into padded X16 (16 x 1024).
__global__ __launch_bounds__(256)
void k_head_pack(const _Float16* __restrict__ Y1, _Float16* __restrict__ X16,
                 int B, int n) {
  const int i = blockIdx.x * blockDim.x + threadIdx.x;
  if (i >= 16 * n) return;
  const int r = i / n, c = i % n;
  X16[i] = (r < B) ? Y1[((size_t)r * n + c) * 16] : (_Float16)0.0f;
}

__global__ __launch_bounds__(256)
void k_cvt_out(const _Float16* __restrict__ src, float* __restrict__ dst, int n) {
  const int i = blockIdx.x * blockDim.x + threadIdx.x;
  if (i < n) dst[i] = (float)src[i];
}

// ---------------------------------------------------------------------------
// Host orchestration
// ---------------------------------------------------------------------------
static inline unsigned cdiv(long long a, long long b) { return (unsigned)((a + b - 1) / b); }

extern "C" void kernel_launch(void* const* d_in, const int* in_sizes, int n_in,
                              void* d_out, int out_size, void* d_ws, size_t ws_size,
                              hipStream_t stream) {
  (void)in_sizes; (void)n_in; (void)out_size; (void)ws_size;
  const int B = 2, N0 = 20000;

  // ---- input mapping (setup_inputs dict insertion order) ----
  const float* PC = (const float*)d_in[0];
  int ip = 1;
  const float *saW[4][3], *saS[4][3], *saB[4][3];
  for (int m = 0; m < 4; ++m) {
    for (int l = 0; l < 3; ++l) saW[m][l] = (const float*)d_in[ip++];
    for (int l = 0; l < 3; ++l) saS[m][l] = (const float*)d_in[ip++];
    for (int l = 0; l < 3; ++l) saB[m][l] = (const float*)d_in[ip++];
  }
  const float *fpW[2][2], *fpS[2][2], *fpB[2][2];
  for (int m = 0; m < 2; ++m) {
    for (int l = 0; l < 2; ++l) fpW[m][l] = (const float*)d_in[ip++];
    for (int l = 0; l < 2; ++l) fpS[m][l] = (const float*)d_in[ip++];
    for (int l = 0; l < 2; ++l) fpB[m][l] = (const float*)d_in[ip++];
  }
  const float* w1w = (const float*)d_in[ip++];
  const float* w1b = (const float*)d_in[ip++];
  const float* w2w = (const float*)d_in[ip++];
  const float* w2b = (const float*)d_in[ip++];

  // ---- workspace bump allocator ----
  size_t off = 0;
  auto alloc = [&](size_t bytes) -> char* {
    off = (off + 255) & ~(size_t)255;
    char* r = (char*)d_ws + off;
    off += bytes;
    return r;
  };

  float* dist = (float*)alloc((size_t)B * N0 * 4);
  const int np[4] = {2048, 1024, 512, 256};
  const int ns[4] = {64, 32, 16, 16};
  const float rad[4] = {0.04f, 0.1f, 0.2f, 0.3f};
  const int cin[4]  = {0, 128, 256, 256};          // prev feature channels
  const int kpad[4] = {32, 160, 288, 288};         // pad(3+cin) to mult of 32
  const int cout[4][3] = {{64, 64, 128}, {128, 128, 256}, {128, 128, 256}, {128, 128, 256}};

  float* nx[4]; _Float16* ft[4];
  for (int m = 0; m < 4; ++m) nx[m] = (float*)alloc((size_t)B * np[m] * 3 * 4);
  for (int m = 0; m < 4; ++m) ft[m] = (_Float16*)alloc((size_t)B * np[m] * cout[m][2] * 2);

  int* idxbuf = (int*)alloc((size_t)B * 2048 * 64 * 4);
  _Float16* Xbuf = (_Float16*)alloc((size_t)B * 32768 * 160 * 2);    // 21 MB max grouped input
  _Float16* ping = (_Float16*)alloc((size_t)B * 131072 * 128 * 2);   // 67 MB
  _Float16* pong = (_Float16*)alloc((size_t)B * 131072 * 128 * 2);   // 67 MB
  _Float16* fpo1 = (_Float16*)alloc((size_t)B * 512 * 256 * 2);
  _Float16* y1   = (_Float16*)alloc((size_t)B * 1024 * 16 * 2);
  _Float16* x16  = (_Float16*)alloc((size_t)16 * 1024 * 2);
  _Float16* y2   = (_Float16*)alloc((size_t)16 * 512 * 2);
  float* b1p  = (float*)alloc(16 * 4);

  // f16 weight storage (zero padded to Npad x Kpad)
  _Float16* saWh[4][3]; _Float16* fpWh[2][2]; _Float16 *w1h, *w2h;
  auto cvt = [&](const float* src, int N, int Npad, int K, int Kpad) -> _Float16* {
    _Float16* dst = (_Float16*)alloc((size_t)Npad * Kpad * 2);
    const int tot = Npad * Kpad;
    k_cvt_w<<<cdiv(tot, 256), 256, 0, stream>>>(src, dst, N, Npad, K, Kpad);
    return dst;
  };
  for (int m = 0; m < 4; ++m) {
    const int k0 = 3 + cin[m];
    saWh[m][0] = cvt(saW[m][0], cout[m][0], cout[m][0], k0, kpad[m]);
    saWh[m][1] = cvt(saW[m][1], cout[m][1], cout[m][1], cout[m][0], cout[m][0]);
    saWh[m][2] = cvt(saW[m][2], cout[m][2], cout[m][2], cout[m][1], cout[m][1]);
  }
  for (int m = 0; m < 2; ++m) {
    fpWh[m][0] = cvt(fpW[m][0], 256, 256, 512, 512);
    fpWh[m][1] = cvt(fpW[m][1], 256, 256, 256, 256);
  }
  w1h = cvt(w1w, 1, 16, 256, 256);
  w2h = cvt(w2w, 512, 512, 1024, 1024);
  k_pad_vec<<<1, 16, 0, stream>>>(w1b, b1p, 1, 16);

  // Pick the big straight-line tile when shapes allow (all SA/FP layers do).
  auto gemm = [&](const _Float16* X, const _Float16* Wp, const float* S,
                  const float* Bv, _Float16* Y, int M, int N, int K) {
    if ((M % 256) == 0 && (N % 64) == 0) {
      dim3 grid((unsigned)(N / 64), (unsigned)(M / 256));
      k_gemm_relu<2, 4><<<grid, 256, 0, stream>>>(X, Wp, S, Bv, Y, M, N, K);
    } else {
      dim3 grid(cdiv(N, 16), cdiv(M, 128));
      k_gemm_relu<1, 1><<<grid, 256, 0, stream>>>(X, Wp, S, Bv, Y, M, N, K);
    }
  };

  // ---- SA stages ----
  const float* xyz_prev = PC;
  const _Float16* ft_prev = nullptr;
  int Nprev = N0;
  for (int m = 0; m < 4; ++m) {
    k_fps<<<B, 1024, 0, stream>>>(xyz_prev, nx[m], dist, Nprev, np[m]);
    const int npts = B * np[m];
    k_ballquery<<<cdiv(npts, 256), 256, 0, stream>>>(
        xyz_prev, nx[m], idxbuf, B, Nprev, np[m], ns[m], rad[m] * rad[m]);
    const int rows = npts * ns[m];
    k_gather<<<cdiv(rows, 256), 256, 0, stream>>>(
        xyz_prev, ft_prev, nx[m], idxbuf, Xbuf, B, Nprev, np[m], ns[m],
        cin[m], kpad[m], 1.0f / rad[m]);
    gemm(Xbuf, saWh[m][0], saS[m][0], saB[m][0], ping, rows, cout[m][0], kpad[m]);
    gemm(ping, saWh[m][1], saS[m][1], saB[m][1], pong, rows, cout[m][1], cout[m][0]);
    gemm(pong, saWh[m][2], saS[m][2], saB[m][2], ping, rows, cout[m][2], cout[m][1]);
    k_maxpool<<<cdiv((long long)npts * cout[m][2], 256), 256, 0, stream>>>(
        ping, ft[m], npts * cout[m][2], ns[m], cout[m][2]);
    xyz_prev = nx[m]; ft_prev = ft[m]; Nprev = np[m];
  }

  // ---- FP stages ----
  k_fp_interp<<<cdiv(B * 512, 256), 256, 0, stream>>>(
      nx[2], nx[3], ft[2], ft[3], Xbuf, B, 512, 256, 256);
  gemm(Xbuf, fpWh[0][0], fpS[0][0], fpB[0][0], ping, B * 512, 256, 512);
  gemm(ping, fpWh[0][1], fpS[0][1], fpB[0][1], fpo1, B * 512, 256, 256);
  k_fp_interp<<<cdiv(B * 1024, 256), 256, 0, stream>>>(
      nx[1], nx[2], ft[1], fpo1, Xbuf, B, 1024, 512, 256);
  gemm(Xbuf, fpWh[1][0], fpS[1][0], fpB[1][0], ping, B * 1024, 256, 512);
  gemm(ping, fpWh[1][1], fpS[1][1], fpB[1][1], pong, B * 1024, 256, 256);

  // ---- heads ----
  gemm(pong, w1h, nullptr, b1p, y1, B * 1024, 16, 256);       // h = relu(f @ w1^T + b)
  k_head_pack<<<cdiv(16 * 1024, 256), 256, 0, stream>>>(y1, x16, B, 1024);
  gemm(x16, w2h, nullptr, w2b, y2, 16, 512, 1024);            // relu(h @ w2^T + b)
  k_cvt_out<<<cdiv(B * 512, 256), 256, 0, stream>>>(y2, (float*)d_out, B * 512);
}
